// SwinSelfAttention_1975684956944
// MI455X (gfx1250) — compile-verified, matrix operand order checked by reading, not run on hardware
//
#include <hip/hip_runtime.h>
#include <hip/hip_bf16.h>
#include <math.h>

// ---------------------------------------------------------------------------
// Swin window self-attention, fully fused, f16 WMMA (gfx1250 / CDNA5, wave32)
//   B=16, NWIN=256 (16x16), NP=64 (8x8 patches), E=512, H=8, HD=64
// ---------------------------------------------------------------------------

typedef __attribute__((ext_vector_type(16))) _Float16 v16h;
typedef __attribute__((ext_vector_type(8)))  float    v8f;

#define E_DIM   512
#define EQKV    1536           // Q|K|V concatenated weight columns
#define NT_ALL  96             // EQKV / 16
#define NP_     64
#define HD_     64
#define XS      520            // LDS stride (halves) for X tile; 520*2 % 16 == 0
#define QS      72             // LDS stride (halves) for 64x64 tiles; 72*2 % 16 == 0
#define SS      65             // LDS stride (floats) for logits

union F16x16 { v16h v; float4 q[2]; };

// Load a 16x32 f16 fragment from row-major storage (LDS or global).
// ISA Sec 7.12.2 wave32 layout: lanes 0-15 -> row = row0+lane, K = {k0..k0+7, k0+16..k0+23}
//                               lanes 16-31 -> row = row0+lane-16, K = {k0+8..15, k0+24..31}
// Used for A fragments, and for B fragments when the B matrix is stored
// transposed (rows indexed by N, contiguous in K).
__device__ inline v16h load_frag_rm(const _Float16* base, int row0, int k0,
                                    int stride, int lane) {
  const int r  = row0 + (lane & 15);
  const int hi = lane >> 4;
  const _Float16* p = base + r * stride + k0 + hi * 8;
  F16x16 u;
  u.q[0] = *(const float4*)(p);
  u.q[1] = *(const float4*)(p + 16);
  return u.v;
}

// Load a pre-swizzled B fragment from global (prep kernel wrote per-lane
// 16-half records): 32 contiguous bytes per lane, fully coalesced per wave.
__device__ inline v16h load_bfrag_swz(const _Float16* __restrict__ wswz,
                                      int kt, int nt, int lane) {
  const _Float16* p = wswz + ((((size_t)kt * NT_ALL + nt) * 32 + lane) << 4);
  F16x16 u;
  u.q[0] = *(const float4*)(p);
  u.q[1] = *(const float4*)(p + 8);
  return u.v;
}

// ---------------------------------------------------------------------------
// Prep: repack Wq|Wk|Wv (fp32 [512,512] each, row = e_in, col = e_out) into
// f16 B-fragment-major layout matching load_bfrag_swz / ISA B layout.
// ---------------------------------------------------------------------------
__global__ void swin_prep_weights(const float* __restrict__ Wq,
                                  const float* __restrict__ Wk,
                                  const float* __restrict__ Wv,
                                  _Float16* __restrict__ wswz) {
  const int i = blockIdx.x * 256 + threadIdx.x;   // 0 .. 512*1536-1
  if (i >= E_DIM * EQKV) return;
  const int e  = i & 15;
  const int L  = (i >> 4) & 31;
  const int f  = i >> 9;            // fragment tile index = kt*NT_ALL + nt
  const int kt = f / NT_ALL;
  const int nt = f - kt * NT_ALL;
  const int hi = L >> 4;
  const int n  = nt * 16 + (L & 15);
  const int ko = (e < 8) ? (hi * 8 + e) : (16 + hi * 8 + (e - 8));
  const int k  = kt * 32 + ko;
  float val;
  if (n < E_DIM)            val = Wq[k * E_DIM + n];
  else if (n < 2 * E_DIM)   val = Wk[k * E_DIM + (n - E_DIM)];
  else                      val = Wv[k * E_DIM + (n - 2 * E_DIM)];
  wswz[i] = (_Float16)val;
}

// ---------------------------------------------------------------------------
// Fused kernel: one 128-thread block per (batch, window).
// ---------------------------------------------------------------------------
__global__ __launch_bounds__(128, 1)
void swin_attn_fused(const float* __restrict__ patches,
                     const float* __restrict__ bq,
                     const float* __restrict__ bk,
                     const float* __restrict__ bv,
                     const float* __restrict__ pos_bias,
                     const _Float16* __restrict__ wswz,
                     float* __restrict__ out) {
  __shared__ _Float16 Xs[64 * XS];       // X tile, f16            (~65 KB)
  __shared__ _Float16 Qs[64 * QS];       // Q head tile  [p][d]
  __shared__ _Float16 Ks[64 * QS];       // K head tile  [q][d]
  __shared__ _Float16 Vt[64 * QS];       // V head tile transposed [d][q]
  __shared__ float    Sf[64 * SS];       // logits f32   [p][q]
  __shared__ _Float16 Pm[64 * QS];       // softmax probs f16 [p][q]
  __shared__ float    pb[240];           // 15x15 relative position bias
  __shared__ float    inv_sum[64];       // per-row softmax 1/sum

  const int bw   = blockIdx.x;           // 0..4095 = b*256 + w
  const int w    = bw & 255;
  const int wy   = w >> 4;
  const int wx   = w & 15;
  const int tid  = threadIdx.x;
  const int lane = tid & 31;
  const int wid  = tid >> 5;             // 4 waves
  const int m0   = wid * 16;             // 16 output rows per wave

  if (tid < 225) pb[tid] = pos_bias[tid];

  // Stage X: 64 rows x 512 f32 -> f16 LDS
  const float* xg = patches + (size_t)bw * (NP_ * E_DIM);
  for (int i = tid; i < 64 * 128; i += 128) {
    const int row  = i >> 7;
    const int col4 = (i & 127) << 2;
    const float4 v = *(const float4*)(xg + row * E_DIM + col4);
    _Float16* d = &Xs[row * XS + col4];
    d[0] = (_Float16)v.x; d[1] = (_Float16)v.y;
    d[2] = (_Float16)v.z; d[3] = (_Float16)v.w;
  }
  __syncthreads();

  const int rbase = ((lane >> 4) << 3);  // 0 or 8: C-matrix row group

  for (int h = 0; h < 8; ++h) {
    // ---------------- QKV projection for this head --------------------
    const int ntQ = h * 4;
    const int ntK = 32 + h * 4;
    const int ntV = 64 + h * 4;
    v8f accq[4] = {}, acck[4] = {}, accv[4] = {};
    for (int kt = 0; kt < 16; ++kt) {
      if (kt < 15)   // prefetch next K-slab of weights (global_prefetch_b8)
        __builtin_prefetch(wswz + ((((size_t)(kt + 1) * NT_ALL + ntQ) * 32 + lane) << 4), 0, 0);
      const v16h a = load_frag_rm(Xs, m0, kt * 32, XS, lane);
#pragma unroll
      for (int t = 0; t < 4; ++t) {
        v16h b = load_bfrag_swz(wswz, kt, ntQ + t, lane);
        accq[t] = __builtin_amdgcn_wmma_f32_16x16x32_f16(false, a, false, b,
                                                         (short)0, accq[t], false, false);
      }
#pragma unroll
      for (int t = 0; t < 4; ++t) {
        v16h b = load_bfrag_swz(wswz, kt, ntK + t, lane);
        acck[t] = __builtin_amdgcn_wmma_f32_16x16x32_f16(false, a, false, b,
                                                         (short)0, acck[t], false, false);
      }
#pragma unroll
      for (int t = 0; t < 4; ++t) {
        v16h b = load_bfrag_swz(wswz, kt, ntV + t, lane);
        accv[t] = __builtin_amdgcn_wmma_f32_16x16x32_f16(false, a, false, b,
                                                         (short)0, accv[t], false, false);
      }
    }
    // Write head tiles to LDS (+bias); V stored transposed for the PV GEMM.
#pragma unroll
    for (int t = 0; t < 4; ++t) {
      const int c   = t * 16 + (lane & 15);
      const float vq = bq[h * HD_ + c];
      const float vk = bk[h * HD_ + c];
      const float vv = bv[h * HD_ + c];
#pragma unroll
      for (int r = 0; r < 8; ++r) {
        const int row = m0 + rbase + r;
        Qs[row * QS + c] = (_Float16)(accq[t][r] + vq);
        Ks[row * QS + c] = (_Float16)(acck[t][r] + vk);
        Vt[c * QS + row] = (_Float16)(accv[t][r] + vv);
      }
    }
    __syncthreads();

    // ---------------- S = Q @ K^T  (64x64, K-dim = 64) ----------------
    {
      v8f accs[4] = {};
#pragma unroll
      for (int kt = 0; kt < 2; ++kt) {
        const v16h a = load_frag_rm(Qs, m0, kt * 32, QS, lane);
#pragma unroll
        for (int t = 0; t < 4; ++t) {
          // B(k=d, n=q) = K[q][d] -> row-major K tile acts as transposed B
          v16h b = load_frag_rm(Ks, t * 16, kt * 32, QS, lane);
          accs[t] = __builtin_amdgcn_wmma_f32_16x16x32_f16(false, a, false, b,
                                                           (short)0, accs[t], false, false);
        }
      }
#pragma unroll
      for (int t = 0; t < 4; ++t) {
        const int c = t * 16 + (lane & 15);
#pragma unroll
        for (int r = 0; r < 8; ++r)
          Sf[(m0 + rbase + r) * SS + c] = accs[t][r];
      }
    }
    __syncthreads();

    // ---------------- softmax with rel-pos bias + shift mask ----------
    if (tid < 64) {
      const int p  = tid;
      const int py = p >> 3, px = p & 7;
      const bool lastRow = (wy == 15), lastCol = (wx == 15);
      float mx = -INFINITY;
      for (int q = 0; q < 64; ++q) {
        const int qy = q >> 3, qx = q & 7;
        float s = Sf[p * SS + q] * 0.125f + pb[(py - qy + 7) * 15 + (px - qx + 7)];
        const bool msk = (lastRow && ((py < 4) != (qy < 4))) ||
                         (lastCol && ((px < 4) != (qx < 4)));
        s = msk ? -INFINITY : s;
        Sf[p * SS + q] = s;
        mx = fmaxf(mx, s);
      }
      float sum = 0.f;
      for (int q = 0; q < 64; ++q) {
        const float e = __expf(Sf[p * SS + q] - mx);
        sum += e;
        Pm[p * QS + q] = (_Float16)e;
      }
      inv_sum[p] = 1.0f / sum;     // folded into output writeback
    }
    __syncthreads();

    // ---------------- O = P @ V  (64x64, K-dim = 64) ------------------
    {
      v8f acco[4] = {};
#pragma unroll
      for (int kt = 0; kt < 2; ++kt) {
        const v16h a = load_frag_rm(Pm, m0, kt * 32, QS, lane);
#pragma unroll
        for (int t = 0; t < 4; ++t) {
          // B(k=q, n=d) = V[q][d] = Vt[d][q] -> Vt rows contiguous in q
          v16h b = load_frag_rm(Vt, t * 16, kt * 32, QS, lane);
          acco[t] = __builtin_amdgcn_wmma_f32_16x16x32_f16(false, a, false, b,
                                                           (short)0, acco[t], false, false);
        }
      }
      float* og = out + (size_t)bw * (NP_ * E_DIM) + h * HD_;
#pragma unroll
      for (int t = 0; t < 4; ++t) {
        const int c = t * 16 + (lane & 15);
#pragma unroll
        for (int r = 0; r < 8; ++r) {
          const int row = m0 + rbase + r;
          og[row * E_DIM + c] = acco[t][r] * inv_sum[row];
        }
      }
    }
    __syncthreads();   // before next head reuses the LDS tiles
  }
}

// ---------------------------------------------------------------------------
// Launch
// ---------------------------------------------------------------------------
extern "C" void kernel_launch(void* const* d_in, const int* in_sizes, int n_in,
                              void* d_out, int out_size, void* d_ws, size_t ws_size,
                              hipStream_t stream) {
  (void)in_sizes; (void)n_in; (void)out_size; (void)ws_size;
  const float* patches  = (const float*)d_in[0];
  const float* Wq       = (const float*)d_in[1];
  const float* bq       = (const float*)d_in[2];
  const float* Wk       = (const float*)d_in[3];
  const float* bk       = (const float*)d_in[4];
  const float* Wv       = (const float*)d_in[5];
  const float* bv       = (const float*)d_in[6];
  const float* pos_bias = (const float*)d_in[7];
  _Float16*    wswz     = (_Float16*)d_ws;     // 512*1536*2 = 1.5 MB scratch
  float*       outp     = (float*)d_out;

  swin_prep_weights<<<(E_DIM * EQKV) / 256, 256, 0, stream>>>(Wq, Wk, Wv, wswz);
  swin_attn_fused<<<16 * 256, 128, 0, stream>>>(patches, bq, bk, bv, pos_bias,
                                                wswz, outp);
}
